// MultiHeadAttention_67791763800130
// MI455X (gfx1250) — compile-verified, matrix operand order checked by reading, not run on hardware
//
#include <hip/hip_runtime.h>
#include <hip/hip_bf16.h>

// ---------- types ----------
typedef _Float16 v16h __attribute__((ext_vector_type(16)));
typedef _Float16 v8h  __attribute__((ext_vector_type(8)));
typedef float    v8f  __attribute__((ext_vector_type(8)));
typedef unsigned int u32x4 __attribute__((ext_vector_type(4)));
typedef int          i32x4 __attribute__((ext_vector_type(4)));
typedef int          i32x8 __attribute__((ext_vector_type(8)));

#define DMODEL 1024
#define SEQ    2048
#define NH     16
#define HD     64
#define NB     4
#define MROWS  (NB * SEQ)   // 8192

#if defined(__has_builtin)
#if __has_builtin(__builtin_amdgcn_tensor_load_to_lds) && __has_builtin(__builtin_amdgcn_s_wait_tensorcnt)
#define HAVE_TDM 1
#endif
#endif
#ifndef HAVE_TDM
#define HAVE_TDM 0
#endif

// A/B-fragment (16x32 f16) load: lane covers row (lane&15), K-chunks split by
// lane half per the CDNA5 16-bit layout (K = lh+0..7 and lh+16..23).
__device__ __forceinline__ v16h ld_frag(const _Float16* __restrict__ p, int lh) {
  union { v16h v; v8h h[2]; } u;
  u.h[0] = *(const v8h*)(p + lh);
  u.h[1] = *(const v8h*)(p + lh + 16);
  return u.v;
}

__device__ __forceinline__ v8f wmma16(v16h a, v16h b, v8f c) {
  return __builtin_amdgcn_wmma_f32_16x16x32_f16(false, a, false, b, (short)0, c,
                                                false, false);
}

#if HAVE_TDM
// Flat generic LDS pointer -> LDS byte offset (low 32 bits of the flat address).
__device__ __forceinline__ unsigned int lds_off(const void* p) {
  return (unsigned int)(uintptr_t)p;
}

// TDM: DMA a rows x 32-half 2D tile (row stride DMODEL halves) Global -> LDS.
// D# fields per CDNA5 ISA:
//   g0: count=1 | lds_addr | global_addr[56:0] | type=2
//   g1: data_size=2B, tensor_dim0/1=1024, tile_dim0=32, tile_dim1=rows,
//       tensor_dim0_stride=1024
__device__ __forceinline__ void tdm_load_tile(const _Float16* gsrc,
                                              unsigned int lds_addr, int rows) {
  unsigned long long ga = (unsigned long long)(uintptr_t)gsrc;
  u32x4 g0;
  g0[0] = 1u;                                   // count=1 (valid), user mode
  g0[1] = lds_addr;                             // LDS byte address
  g0[2] = (unsigned int)ga;                     // global_addr[31:0]
  g0[3] = (unsigned int)(ga >> 32) | 0x80000000u; // global_addr[56:32] | type=2
  i32x8 g1;
  g1[0] = (int)(1u << 16);                      // data_size code 1 = 2 bytes
  g1[1] = (int)((DMODEL & 0xFFFFu) << 16);      // tensor_dim0[15:0]
  g1[2] = (int)((DMODEL >> 16) | ((DMODEL & 0xFFFFu) << 16)); // dim0 hi | dim1 lo
  g1[3] = (int)((DMODEL >> 16) | (32u << 16));  // dim1 hi | tile_dim0 = 32
  g1[4] = rows;                                 // tile_dim1 | tile_dim2 = 0
  g1[5] = DMODEL;                               // tensor_dim0_stride[31:0]
  g1[6] = 0;                                    // stride0 hi | dim1_stride lo
  g1[7] = 0;
  i32x4 z4 = {0, 0, 0, 0};
#if defined(__clang_major__) && __clang_major__ >= 23
  i32x8 z8 = {0, 0, 0, 0, 0, 0, 0, 0};
  __builtin_amdgcn_tensor_load_to_lds(g0, g1, z4, z4, z8, 0);
#else
  __builtin_amdgcn_tensor_load_to_lds(g0, g1, z4, z4, 0);
#endif
}
#endif // HAVE_TDM

// ---------- fp32 -> fp16 convert ----------
__global__ void cvt_f32_f16(const float* __restrict__ src,
                            _Float16* __restrict__ dst, int n) {
  int i = blockIdx.x * blockDim.x + threadIdx.x;
  int stride = gridDim.x * blockDim.x;
  for (; i < n; i += stride) dst[i] = (_Float16)src[i];
}

// ---------- shared GEMM core ----------
// Block tile 128(M) x 64(N); 4 waves, wave w owns rows w*32..w*32+31 as two
// 16-row subtiles sharing the four B fragments (8 WMMA / 12 ds_load per step).
// A/B tiles staged in LDS by the Tensor Data Mover, double-buffered: DMA for
// step k+32 overlaps the WMMAs of step k; B fetched once per workgroup.
// All fragments of a step are loaded before any WMMA so the matrix ops issue
// back-to-back behind a single s_wait_dscnt.
__device__ __forceinline__ void gemm_core(const _Float16* __restrict__ A,
                                          const _Float16* __restrict__ W,
                                          int mBase, int nBase,
                                          _Float16 (*Alds)[128][32],
                                          _Float16 (*Blds)[64][32],
                                          v8f (*acc)[4]) {
  const int tid  = threadIdx.x;
  const int lane = tid & 31;
  const int wid  = tid >> 5;
  const int lm   = lane & 15;
  const int lh   = (lane < 16) ? 0 : 8;
  const int r0   = wid * 32 + lm;

#if HAVE_TDM
  int cur = 0;
  if (wid == 0) {
    tdm_load_tile(A + (size_t)mBase * DMODEL, lds_off(&Alds[0][0][0]), 128);
    tdm_load_tile(W + (size_t)nBase * DMODEL, lds_off(&Blds[0][0][0]), 64);
  }
  for (int kk = 0; kk < DMODEL; kk += 32) {
    if (wid == 0) __builtin_amdgcn_s_wait_tensorcnt((unsigned short)0);
    __syncthreads();                       // buffer `cur` ready; prev reads done
    if (wid == 0 && (kk + 32) < DMODEL) {  // DMA next tiles into other buffer
      tdm_load_tile(A + (size_t)mBase * DMODEL + kk + 32,
                    lds_off(&Alds[cur ^ 1][0][0]), 128);
      tdm_load_tile(W + (size_t)nBase * DMODEL + kk + 32,
                    lds_off(&Blds[cur ^ 1][0][0]), 64);
    }
    v16h a0 = ld_frag(&Alds[cur][r0][0], lh);
    v16h a1 = ld_frag(&Alds[cur][r0 + 16][0], lh);
    v16h b0 = ld_frag(&Blds[cur][lm][0], lh);
    v16h b1 = ld_frag(&Blds[cur][16 + lm][0], lh);
    v16h b2 = ld_frag(&Blds[cur][32 + lm][0], lh);
    v16h b3 = ld_frag(&Blds[cur][48 + lm][0], lh);
    acc[0][0] = wmma16(a0, b0, acc[0][0]);
    acc[0][1] = wmma16(a0, b1, acc[0][1]);
    acc[0][2] = wmma16(a0, b2, acc[0][2]);
    acc[0][3] = wmma16(a0, b3, acc[0][3]);
    acc[1][0] = wmma16(a1, b0, acc[1][0]);
    acc[1][1] = wmma16(a1, b1, acc[1][1]);
    acc[1][2] = wmma16(a1, b2, acc[1][2]);
    acc[1][3] = wmma16(a1, b3, acc[1][3]);
    cur ^= 1;
  }
#else
  for (int kk = 0; kk < DMODEL; kk += 32) {
    __syncthreads();
    for (int i = tid; i < 128 * 4; i += 128) {  // A: 128 rows x 4 chunks
      int r = i >> 2, c = (i & 3) * 8;
      *(v8h*)&Alds[0][r][c] = *(const v8h*)(A + (size_t)(mBase + r) * DMODEL + kk + c);
    }
    for (int i = tid; i < 64 * 4; i += 128) {   // B: 64 rows x 4 chunks
      int r = i >> 2, c = (i & 3) * 8;
      *(v8h*)&Blds[0][r][c] = *(const v8h*)(W + (size_t)(nBase + r) * DMODEL + kk + c);
    }
    __syncthreads();
    v16h a0 = ld_frag(&Alds[0][r0][0], lh);
    v16h a1 = ld_frag(&Alds[0][r0 + 16][0], lh);
    v16h b0 = ld_frag(&Blds[0][lm][0], lh);
    v16h b1 = ld_frag(&Blds[0][16 + lm][0], lh);
    v16h b2 = ld_frag(&Blds[0][32 + lm][0], lh);
    v16h b3 = ld_frag(&Blds[0][48 + lm][0], lh);
    acc[0][0] = wmma16(a0, b0, acc[0][0]);
    acc[0][1] = wmma16(a0, b1, acc[0][1]);
    acc[0][2] = wmma16(a0, b2, acc[0][2]);
    acc[0][3] = wmma16(a0, b3, acc[0][3]);
    acc[1][0] = wmma16(a1, b0, acc[1][0]);
    acc[1][1] = wmma16(a1, b1, acc[1][1]);
    acc[1][2] = wmma16(a1, b2, acc[1][2]);
    acc[1][3] = wmma16(a1, b3, acc[1][3]);
  }
#endif
}

// ---------- QKV projection: mode 0 -> [b,h,s,d] f16, mode 1 -> [b,h,d,s] f16 ----------
__global__ void gemm_qkv(const _Float16* __restrict__ A,
                         const _Float16* __restrict__ W,
                         _Float16* __restrict__ out, int mode) {
  __shared__ __align__(16) _Float16 Alds[2][128][32];
  __shared__ __align__(16) _Float16 Blds[2][64][32];
  const int lane  = threadIdx.x & 31;
  const int wid   = threadIdx.x >> 5;
  const int nBase = blockIdx.x * 64;
  const int lm    = lane & 15;
  const int hlf   = lane >> 4;

  v8f acc[2][4] = {};
  gemm_core(A, W, blockIdx.y * 128, nBase, Alds, Blds, acc);

#pragma unroll
  for (int sub = 0; sub < 2; ++sub) {
    const int mrow = blockIdx.y * 128 + wid * 32 + sub * 16;
#pragma unroll
    for (int t = 0; t < 4; ++t) {
      const int col = nBase + t * 16 + lm;
      const int h = col >> 6, dd = col & 63;
#pragma unroll
      for (int r = 0; r < 8; ++r) {
        const int g  = mrow + r + 8 * hlf;        // global row (b*SEQ+s)
        const int bi = g >> 11, s = g & (SEQ - 1);
        _Float16 v = (_Float16)acc[sub][t][r];
        if (mode == 0)
          out[((size_t)(bi * NH + h) * SEQ + s) * HD + dd] = v;
        else
          out[((size_t)(bi * NH + h) * HD + dd) * SEQ + s] = v;
      }
    }
  }
}

// ---------- out = ctx @ Wo^T + bo, fp32 output ----------
__global__ void gemm_out(const _Float16* __restrict__ A,
                         const _Float16* __restrict__ W,
                         const float* __restrict__ bias,
                         float* __restrict__ out) {
  __shared__ __align__(16) _Float16 Alds[2][128][32];
  __shared__ __align__(16) _Float16 Blds[2][64][32];
  const int lane  = threadIdx.x & 31;
  const int wid   = threadIdx.x >> 5;
  const int nBase = blockIdx.x * 64;
  const int lm    = lane & 15;
  const int hlf   = lane >> 4;

  v8f acc[2][4] = {};
  gemm_core(A, W, blockIdx.y * 128, nBase, Alds, Blds, acc);

#pragma unroll
  for (int sub = 0; sub < 2; ++sub) {
    const int mrow = blockIdx.y * 128 + wid * 32 + sub * 16;
#pragma unroll
    for (int t = 0; t < 4; ++t) {
      const int col = nBase + t * 16 + lm;
      const float bb = bias[col];
#pragma unroll
      for (int r = 0; r < 8; ++r) {
        const int g = mrow + r + 8 * hlf;
        out[(size_t)g * DMODEL + col] = acc[sub][t][r] + bb;
      }
    }
  }
}

// ---------- flash attention: 1 block = (b,h,64 query rows); 4 waves x 16 rows ----------
__global__ void attn_flash(const _Float16* __restrict__ Q,
                           const _Float16* __restrict__ K,
                           const _Float16* __restrict__ Vt,
                           _Float16* __restrict__ ctx) {
  __shared__ __align__(16) _Float16 Plds[4][16][32];
  const int lane = threadIdx.x & 31;
  const int wid  = threadIdx.x >> 5;
  const int qt = blockIdx.x, h = blockIdx.y, bi = blockIdx.z;
  const int qbase = qt * 64 + wid * 16;
  const size_t bh = (size_t)(bi * NH + h);
  const _Float16* Qp = Q  + bh * SEQ * HD;
  const _Float16* Kp = K  + bh * SEQ * HD;
  const _Float16* Vp = Vt + bh * HD * SEQ;
  const int lm  = lane & 15;
  const int lh  = (lane < 16) ? 0 : 8;
  const int hlf = lane >> 4;

  v16h aq[2];
  aq[0] = ld_frag(Qp + (size_t)(qbase + lm) * HD, lh);
  aq[1] = ld_frag(Qp + (size_t)(qbase + lm) * HD + 32, lh);

  float M8[8], L8[8];
  v8f acc[4] = {};
#pragma unroll
  for (int r = 0; r < 8; ++r) { M8[r] = -1e30f; L8[r] = 0.f; }

  const int nblocks = (qbase + 16 + 31) >> 5;   // keys < qbase+16, 32-key blocks
  for (int jb = 0; jb < nblocks; ++jb) {
    const int j = jb * 32;
    // scores: S = Q (16x64) @ K_block^T  -> two 16x16 tiles
    // (all fragments loaded first so the 4 WMMAs issue back-to-back)
    v16h bk0 = ld_frag(Kp + (size_t)(j + lm) * HD, lh);
    v16h bk1 = ld_frag(Kp + (size_t)(j + lm) * HD + 32, lh);
    v16h bk2 = ld_frag(Kp + (size_t)(j + 16 + lm) * HD, lh);
    v16h bk3 = ld_frag(Kp + (size_t)(j + 16 + lm) * HD + 32, lh);
    v8f sc[2] = {};
    sc[0] = wmma16(aq[0], bk0, sc[0]);
    sc[0] = wmma16(aq[1], bk1, sc[0]);
    sc[1] = wmma16(aq[0], bk2, sc[1]);
    sc[1] = wmma16(aq[1], bk3, sc[1]);
    // scale (1/sqrt(64)) + causal mask
#pragma unroll
    for (int t = 0; t < 2; ++t) {
      const int key = j + t * 16 + lm;
#pragma unroll
      for (int r = 0; r < 8; ++r) {
        const int qr = qbase + r + 8 * hlf;
        float v = sc[t][r] * 0.125f;
        sc[t][r] = (key <= qr) ? v : -1e30f;
      }
    }
    // online softmax (row stats live per lane-half; xor masks <16 stay in half)
    float alpha[8];
#pragma unroll
    for (int r = 0; r < 8; ++r) {
      float mx = fmaxf(sc[0][r], sc[1][r]);
#pragma unroll
      for (int o = 8; o >= 1; o >>= 1) mx = fmaxf(mx, __shfl_xor(mx, o, 32));
      const float nm = fmaxf(M8[r], mx);
      alpha[r] = __expf(M8[r] - nm);
      float p0 = __expf(sc[0][r] - nm);
      float p1 = __expf(sc[1][r] - nm);
      sc[0][r] = p0; sc[1][r] = p1;
      float rs = p0 + p1;
#pragma unroll
      for (int o = 8; o >= 1; o >>= 1) rs += __shfl_xor(rs, o, 32);
      L8[r] = L8[r] * alpha[r] + rs;
      M8[r] = nm;
    }
#pragma unroll
    for (int t2 = 0; t2 < 4; ++t2)
#pragma unroll
      for (int r = 0; r < 8; ++r) acc[t2][r] *= alpha[r];

    // P tile: C-layout -> LDS -> reload as A-fragment (per-wave region;
    // DS ops are in-order per wave; explicit CDNA5 split-counter wait since
    // waves have divergent causal trip counts and cannot block-barrier here)
#pragma unroll
    for (int t = 0; t < 2; ++t)
#pragma unroll
      for (int r = 0; r < 8; ++r)
        Plds[wid][r + 8 * hlf][t * 16 + lm] = (_Float16)sc[t][r];
    asm volatile("s_wait_dscnt 0x0" ::: "memory");
    union { v16h v; v8h h2[2]; } up;
    up.h2[0] = *(const v8h*)&Plds[wid][lm][lh];
    up.h2[1] = *(const v8h*)&Plds[wid][lm][lh + 16];

    // ctx += P (16x32) @ V_block (32x64); Vt rows are key-contiguous
    v16h bv0 = ld_frag(Vp + (size_t)(0 * 16 + lm) * SEQ + j, lh);
    v16h bv1 = ld_frag(Vp + (size_t)(1 * 16 + lm) * SEQ + j, lh);
    v16h bv2 = ld_frag(Vp + (size_t)(2 * 16 + lm) * SEQ + j, lh);
    v16h bv3 = ld_frag(Vp + (size_t)(3 * 16 + lm) * SEQ + j, lh);
    acc[0] = wmma16(up.v, bv0, acc[0]);
    acc[1] = wmma16(up.v, bv1, acc[1]);
    acc[2] = wmma16(up.v, bv2, acc[2]);
    acc[3] = wmma16(up.v, bv3, acc[3]);
  }

  float inv[8];
#pragma unroll
  for (int r = 0; r < 8; ++r) inv[r] = 1.0f / L8[r];
#pragma unroll
  for (int t2 = 0; t2 < 4; ++t2) {
    const int col = h * HD + t2 * 16 + lm;
#pragma unroll
    for (int r = 0; r < 8; ++r) {
      const int qr = qbase + r + 8 * hlf;
      ctx[((size_t)(bi * SEQ + qr)) * DMODEL + col] = (_Float16)(acc[t2][r] * inv[r]);
    }
  }
}

// ---------- host ----------
extern "C" void kernel_launch(void* const* d_in, const int* in_sizes, int n_in,
                              void* d_out, int out_size, void* d_ws, size_t ws_size,
                              hipStream_t stream) {
  (void)in_sizes; (void)n_in; (void)out_size; (void)ws_size;
  const float* x  = (const float*)d_in[0];
  const float* Wq = (const float*)d_in[1];
  const float* Wk = (const float*)d_in[2];
  const float* Wv = (const float*)d_in[3];
  const float* Wo = (const float*)d_in[4];
  const float* bo = (const float*)d_in[5];

  _Float16* ws = (_Float16*)d_ws;
  size_t off = 0;
  _Float16* xh  = ws + off; off += (size_t)MROWS * DMODEL;   // 16 MB
  _Float16* wqh = ws + off; off += (size_t)DMODEL * DMODEL;  // 2 MB each
  _Float16* wkh = ws + off; off += (size_t)DMODEL * DMODEL;
  _Float16* wvh = ws + off; off += (size_t)DMODEL * DMODEL;
  _Float16* woh = ws + off; off += (size_t)DMODEL * DMODEL;
  _Float16* qh  = ws + off; off += (size_t)MROWS * DMODEL;   // [b,h,s,d]
  _Float16* kh  = ws + off; off += (size_t)MROWS * DMODEL;   // [b,h,s,d]
  _Float16* vth = ws + off; off += (size_t)MROWS * DMODEL;   // [b,h,d,s]
  _Float16* cth = ws + off; off += (size_t)MROWS * DMODEL;   // [b,s,D]

  const int nx = MROWS * DMODEL, nw = DMODEL * DMODEL;
  cvt_f32_f16<<<(nx + 255) / 256, 256, 0, stream>>>(x,  xh,  nx);
  cvt_f32_f16<<<(nw + 255) / 256, 256, 0, stream>>>(Wq, wqh, nw);
  cvt_f32_f16<<<(nw + 255) / 256, 256, 0, stream>>>(Wk, wkh, nw);
  cvt_f32_f16<<<(nw + 255) / 256, 256, 0, stream>>>(Wv, wvh, nw);
  cvt_f32_f16<<<(nw + 255) / 256, 256, 0, stream>>>(Wo, woh, nw);

  dim3 gg(DMODEL / 64, MROWS / 128);           // (16, 64)
  gemm_qkv<<<gg, 128, 0, stream>>>(xh, wqh, qh, 0);
  gemm_qkv<<<gg, 128, 0, stream>>>(xh, wkh, kh, 0);
  gemm_qkv<<<gg, 128, 0, stream>>>(xh, wvh, vth, 1);

  attn_flash<<<dim3(SEQ / 64, NH, NB), 128, 0, stream>>>(qh, kh, vth, cth);

  gemm_out<<<gg, 128, 0, stream>>>(cth, woh, bo, (float*)d_out);
}
